// HopfieldAttention_73744588472843
// MI455X (gfx1250) — matrix operand order repaired and checked by reading, not compiled, not run on hardware
//
#include <hip/hip_runtime.h>
#include <hip/hip_bf16.h>

typedef __attribute__((ext_vector_type(16))) __bf16 v16bf;
typedef __attribute__((ext_vector_type(8)))  float  v8f;

constexpr int Bn = 2;       // batch
constexpr int Sq = 2048;    // sequence
constexpr int Dm = 1024;    // model dim
constexpr int Hn = 16;      // heads
constexpr int Dh = 64;      // head dim
constexpr int Mrows = Bn * Sq;  // 4096

union AFrag { uint4 q[2]; v16bf v; };

__device__ __forceinline__ unsigned short f2bf(float f) {
  unsigned u = __float_as_uint(f);
  unsigned r = u + 0x7FFFu + ((u >> 16) & 1u);   // round-to-nearest-even
  return (unsigned short)(r >> 16);
}
__device__ __forceinline__ unsigned pack2(float a, float b) {
  return (unsigned)f2bf(a) | ((unsigned)f2bf(b) << 16);
}

// DPP cross-lane fetch (within a 16-lane DPP row); pure VALU, no LDS.
template <int CTRL>
__device__ __forceinline__ float dpp_shuf(float v) {
  return __int_as_float(
      __builtin_amdgcn_update_dpp(0, __float_as_int(v), CTRL, 0xF, 0xF, true));
}
// 16-lane all-reduce max: xor1, xor2, half-mirror (pairs 8-lane halves),
// mirror (pairs 16-lane halves). 4 DPP+max ops, no ds_bpermute round-trips.
__device__ __forceinline__ float red_max16(float v) {
  v = fmaxf(v, dpp_shuf<0xB1>(v));    // quad_perm [1,0,3,2]
  v = fmaxf(v, dpp_shuf<0x4E>(v));    // quad_perm [2,3,0,1]
  v = fmaxf(v, dpp_shuf<0x141>(v));   // row_half_mirror
  v = fmaxf(v, dpp_shuf<0x140>(v));   // row_mirror
  return v;
}

// A fragment (16x32 bf16): lane L holds row M=L%16; halves j0..7 -> K = 8*(L/16)+0..7,
// j8..15 -> K = 8*(L/16)+16..23  => two contiguous 16B chunks at +0 and +16 halfwords.
__device__ __forceinline__ v16bf ld_frag_a(const unsigned short* lds, int rowBase,
                                           int stride, int kbase, int lane) {
  int row = rowBase + (lane & 15);
  int off = (lane >> 4) << 3;                    // 0 or 8
  const unsigned short* p = lds + row * stride + kbase + off;
  AFrag f;
  f.q[0] = *(const uint4*)(p);
  f.q[1] = *(const uint4*)(p + 16);
  return f.v;
}
// B fragment (32x16 bf16), LDS stored [n][k]: lane L holds col N=L%16;
// halves j0..7 -> K = 8*(L/16)+0..7, j8..15 -> K = 8*(L/16)+16..23 => two 16B chunks.
__device__ __forceinline__ v16bf ld_frag_b(const unsigned short* lds, int nBase,
                                           int stride, int kbase, int lane) {
  int n = nBase + (lane & 15);
  int off = (lane >> 4) << 3;                    // 0 or 8
  const unsigned short* p = lds + n * stride + kbase + off;
  AFrag f;
  f.q[0] = *(const uint4*)(p);
  f.q[1] = *(const uint4*)(p + 16);
  return f.v;
}

// ---------------------------------------------------------------------------
// GEMM: C[M,N] = A[M,K] * B[K,N] + bias[N].  A fp32 or bf16; B/bias fp32.
// 128x128 tile per 256-thread block, K stepped by 32 through LDS, bf16 WMMA.
// ---------------------------------------------------------------------------
template <bool A_BF16>
__global__ void __launch_bounds__(256)
gemm_bias_kernel(const void* __restrict__ Aptr, const float* __restrict__ Bw,
                 const float* __restrict__ bias, float* __restrict__ C,
                 int M, int N, int K) {
  __shared__ __align__(16) unsigned short As[128 * 40];
  __shared__ __align__(16) unsigned short Bs[128 * 40];

  const int tid  = threadIdx.x;
  const int lane = tid & 31;
  const int wave = tid >> 5;
  const int bm = blockIdx.y * 128;
  const int bn = blockIdx.x * 128;
  const int wr = (wave >> 1) * 32;   // 0,32,64,96
  const int wc = (wave & 1) * 64;    // 0,64

  v8f acc[2][4];
  const v8f vz = {0.f, 0.f, 0.f, 0.f, 0.f, 0.f, 0.f, 0.f};
#pragma unroll
  for (int mi = 0; mi < 2; ++mi)
#pragma unroll
    for (int ni = 0; ni < 4; ++ni) acc[mi][ni] = vz;

  for (int k0 = 0; k0 < K; k0 += 32) {
    // ---- stage A slice: 128x32 -> As[m][k] (bf16)
    {
      int row = tid >> 1;
      int cb  = (tid & 1) * 16;
      unsigned short* dst = &As[row * 40 + cb];
      if (A_BF16) {
        const unsigned short* g =
            (const unsigned short*)Aptr + (size_t)(bm + row) * K + k0 + cb;
        ((uint4*)dst)[0] = ((const uint4*)g)[0];
        ((uint4*)dst)[1] = ((const uint4*)g)[1];
        if (k0 + 32 < K) __builtin_prefetch(g + 32, 0, 1);
      } else {
        const float* g = (const float*)Aptr + (size_t)(bm + row) * K + k0 + cb;
        float4 f0 = ((const float4*)g)[0];
        float4 f1 = ((const float4*)g)[1];
        float4 f2 = ((const float4*)g)[2];
        float4 f3 = ((const float4*)g)[3];
        unsigned* d32 = (unsigned*)dst;
        d32[0] = pack2(f0.x, f0.y); d32[1] = pack2(f0.z, f0.w);
        d32[2] = pack2(f1.x, f1.y); d32[3] = pack2(f1.z, f1.w);
        d32[4] = pack2(f2.x, f2.y); d32[5] = pack2(f2.z, f2.w);
        d32[6] = pack2(f3.x, f3.y); d32[7] = pack2(f3.z, f3.w);
        if (k0 + 32 < K) __builtin_prefetch(g + 32, 0, 1);
      }
    }
    // ---- stage B slice: 32x128 transposed -> Bs[n][k] (bf16)
    {
      int kk = tid & 31;
      int ng = tid >> 5;  // 0..7 -> 16 cols each
      const float* g = Bw + (size_t)(k0 + kk) * N + bn + ng * 16;
      float tv[16];
      ((float4*)tv)[0] = ((const float4*)g)[0];
      ((float4*)tv)[1] = ((const float4*)g)[1];
      ((float4*)tv)[2] = ((const float4*)g)[2];
      ((float4*)tv)[3] = ((const float4*)g)[3];
      if (k0 + 32 < K) __builtin_prefetch(g + (size_t)32 * N, 0, 1);
#pragma unroll
      for (int j = 0; j < 16; ++j)
        Bs[(ng * 16 + j) * 40 + kk] = f2bf(tv[j]);
    }
    __syncthreads();

    v16bf af[2], bf[4];
#pragma unroll
    for (int mi = 0; mi < 2; ++mi) af[mi] = ld_frag_a(As, wr + mi * 16, 40, 0, lane);
#pragma unroll
    for (int ni = 0; ni < 4; ++ni) bf[ni] = ld_frag_b(Bs, wc + ni * 16, 40, 0, lane);
#pragma unroll
    for (int mi = 0; mi < 2; ++mi)
#pragma unroll
      for (int ni = 0; ni < 4; ++ni)
        acc[mi][ni] = __builtin_amdgcn_wmma_f32_16x16x32_bf16(
            false, af[mi], false, bf[ni], (short)0, acc[mi][ni], false, false);
    __syncthreads();
  }

  // epilogue: C layout VGPR i -> row i + 8*(lane/16), col = lane%16
  const int half = lane >> 4;
  const int nl   = lane & 15;
#pragma unroll
  for (int mi = 0; mi < 2; ++mi)
#pragma unroll
    for (int ni = 0; ni < 4; ++ni)
#pragma unroll
      for (int i = 0; i < 8; ++i) {
        int row = bm + wr + mi * 16 + i + 8 * half;
        int col = bn + wc + ni * 16 + nl;
        C[(size_t)row * N + col] = acc[mi][ni][i] + bias[col];
      }
}

// ---------------------------------------------------------------------------
// LayerNorm (per 64-elem head row) + repack fp32 (b,s,h*64) -> bf16 (b,h,s,64)
// One wave per row; 8 rows per 256-thread block.  outScale folds the softmax
// logit scale (1/sqrt(d) * log2(e)) into the Q path.
// ---------------------------------------------------------------------------
__global__ void __launch_bounds__(256)
ln_pack_kernel(const float* __restrict__ X, const float* __restrict__ gamma,
               const float* __restrict__ beta, unsigned short* __restrict__ Y,
               int doLN, float outScale) {
  int r    = blockIdx.x * 8 + (threadIdx.x >> 5);   // (b,h,s) flat
  int lane = threadIdx.x & 31;
  int b = r >> 15;            // / (Hn*Sq) = 32768
  int h = (r >> 11) & 15;     // / Sq % Hn
  int s = r & 2047;
  const float* x = X + ((size_t)(b * Sq + s) * Dm) + h * Dh;
  float v0 = x[lane * 2 + 0];
  float v1 = x[lane * 2 + 1];
  if (doLN) {
    float sum = v0 + v1, sq = v0 * v0 + v1 * v1;
#pragma unroll
    for (int m = 16; m >= 1; m >>= 1) {
      sum += __shfl_xor(sum, m, 32);
      sq  += __shfl_xor(sq,  m, 32);
    }
    float mu  = sum * (1.f / 64.f);
    float var = sq * (1.f / 64.f) - mu * mu;
    float rs  = rsqrtf(var + 1e-5f);
    v0 = (v0 - mu) * rs * gamma[lane * 2 + 0] + beta[lane * 2 + 0];
    v1 = (v1 - mu) * rs * gamma[lane * 2 + 1] + beta[lane * 2 + 1];
  }
  v0 *= outScale;
  v1 *= outScale;
  unsigned short* y = Y + (size_t)r * Dh;
  *(unsigned*)&y[lane * 2] = pack2(v0, v1);
}

// ---------------------------------------------------------------------------
// Flash-style attention: O = softmax2(Q Kt) @ V, per (b,h), 128 q-rows/block.
// Q is pre-scaled by 1/sqrt(d)*log2(e), so softmax runs in the exp2 domain.
// 256 threads (8 waves); wave w owns score rows [w*16, w*16+16) across all N.
// Row-max reduction: DPP (no LDS). Row-sum: accumulated by WMMA against a
// constant ones-column B fragment (l transforms identically to O).
// ---------------------------------------------------------------------------
__global__ void __launch_bounds__(256)
hopfield_attn_kernel(const unsigned short* __restrict__ Q,
                     const unsigned short* __restrict__ Kh,
                     const unsigned short* __restrict__ Vh,
                     unsigned short* __restrict__ O,
                     long long oHStr, long long oSStr, float outScale) {
  __shared__ __align__(16) unsigned short Qs[128 * 72];
  __shared__ __align__(16) unsigned short Ks[64 * 72];
  __shared__ __align__(16) unsigned short Vs[64 * 72];  // [d][key]
  __shared__ __align__(16) unsigned short Ps[128 * 72];

  const int qt = blockIdx.x, h = blockIdx.y, b = blockIdx.z;
  const int tid = threadIdx.x, lane = tid & 31, wave = tid >> 5;
  const size_t bh = ((size_t)b * Hn + h) * Sq;
  const int qbase = qt * 128;
  const int half = lane >> 4, nl = lane & 15;

  // load Q tile (128x64) -> Qs natural row-major
  {
    int row = tid >> 1;
    int c   = (tid & 1) * 32;
    const uint4* g = (const uint4*)(Q + (bh + qbase + row) * Dh + c);
    uint4* d = (uint4*)&Qs[row * 72 + c];
    d[0] = g[0]; d[1] = g[1]; d[2] = g[2]; d[3] = g[3];
  }
  __syncthreads();
  // Q A-fragments are invariant across the whole key stream: hoist them.
  v16bf aq[2];
  aq[0] = ld_frag_a(Qs, wave * 16, 72, 0, lane);
  aq[1] = ld_frag_a(Qs, wave * 16, 72, 32, lane);

  // Constant ones-column B fragment: B[k][0] = 1 for all k, other cols 0.
  // Lane layout: lane nl==0 (lanes 0,16) holds column 0 across all 16 K-halves.
  v16bf bones;
  {
    AFrag fb;
    unsigned fill = (nl == 0) ? 0x3F803F80u : 0u;   // two bf16 1.0s
    fb.q[0].x = fill; fb.q[0].y = fill; fb.q[0].z = fill; fb.q[0].w = fill;
    fb.q[1] = fb.q[0];
    bones = fb.v;
  }

  float m[8];
  v8f o[4], osum;
  const v8f vz = {0.f, 0.f, 0.f, 0.f, 0.f, 0.f, 0.f, 0.f};
#pragma unroll
  for (int i = 0; i < 8; ++i) m[i] = -__builtin_inff();
#pragma unroll
  for (int t = 0; t < 4; ++t) o[t] = vz;
  osum = vz;

  for (int kc = 0; kc < Sq; kc += 64) {
    __syncthreads();  // previous iteration's LDS consumers done
    // K chunk -> Ks natural [key][d]; prefetch next chunk into L2
    {
      int row = tid >> 2;
      int c   = (tid & 3) * 16;
      const unsigned short* gk = Kh + (bh + kc + row) * Dh + c;
      uint4* d = (uint4*)&Ks[row * 72 + c];
      d[0] = ((const uint4*)gk)[0];
      d[1] = ((const uint4*)gk)[1];
      if (kc + 64 < Sq) __builtin_prefetch(gk + (size_t)64 * Dh, 0, 1);
    }
    // V chunk transposed -> Vs[d][key]
    {
      int key = tid >> 2;
      int db  = (tid & 3) * 16;
      const unsigned short* gv = Vh + (bh + kc + key) * Dh + db;
      uint4 u[2] = {((const uint4*)gv)[0], ((const uint4*)gv)[1]};
      const unsigned short* us = (const unsigned short*)u;
#pragma unroll
      for (int i = 0; i < 16; ++i) Vs[(db + i) * 72 + key] = us[i];
      if (kc + 64 < Sq) __builtin_prefetch(gv + (size_t)64 * Dh, 0, 1);
    }
    __syncthreads();

    // ---- scores: S = Q Kt (128x64), each wave computes 16 rows x 64 cols
    v8f sc[4];
#pragma unroll
    for (int t = 0; t < 4; ++t) sc[t] = vz;
#pragma unroll
    for (int kk = 0; kk < 2; ++kk) {
      v16bf bb[4];
#pragma unroll
      for (int t = 0; t < 4; ++t) bb[t] = ld_frag_b(Ks, t * 16, 72, kk * 32, lane);
#pragma unroll
      for (int t = 0; t < 4; ++t)
        sc[t] = __builtin_amdgcn_wmma_f32_16x16x32_bf16(
            false, aq[kk], false, bb[t], (short)0, sc[t], false, false);
    }

    // ---- online softmax in exp2 domain; row max via DPP all-reduce
    float alpha[8];
#pragma unroll
    for (int i = 0; i < 8; ++i) {
      float mc = fmaxf(fmaxf(sc[0][i], sc[1][i]), fmaxf(sc[2][i], sc[3][i]));
      mc = red_max16(mc);
      float mn = fmaxf(m[i], mc);
      alpha[i] = __builtin_exp2f(m[i] - mn);
      m[i] = mn;
    }
#pragma unroll
    for (int t = 0; t < 4; ++t)
#pragma unroll
      for (int i = 0; i < 8; ++i)
        sc[t][i] = __builtin_exp2f(sc[t][i] - m[i]);
#pragma unroll
    for (int t = 0; t < 4; ++t)
#pragma unroll
      for (int i = 0; i < 8; ++i) o[t][i] *= alpha[i];
#pragma unroll
    for (int i = 0; i < 8; ++i) osum[i] *= alpha[i];

    // ---- P (bf16) -> Ps; each wave writes exactly its own 16 rows
#pragma unroll
    for (int t = 0; t < 4; ++t)
#pragma unroll
      for (int i = 0; i < 8; ++i) {
        int row = wave * 16 + i + 8 * half;
        Ps[row * 72 + t * 16 + nl] = f2bf(sc[t][i]);
      }
    // no barrier needed: PV A-fragments read only this wave's rows of Ps

    // ---- O += P @ V ; l += P @ ones  (M=16 rows/wave, N=64(+1), K=64 keys)
#pragma unroll
    for (int kk = 0; kk < 2; ++kk) {
      v16bf ap = ld_frag_a(Ps, wave * 16, 72, kk * 32, lane);
      v16bf bb[4];
#pragma unroll
      for (int t = 0; t < 4; ++t) bb[t] = ld_frag_b(Vs, t * 16, 72, kk * 32, lane);
#pragma unroll
      for (int t = 0; t < 4; ++t)
        o[t] = __builtin_amdgcn_wmma_f32_16x16x32_bf16(
            false, ap, false, bb[t], (short)0, o[t], false, false);
      osum = __builtin_amdgcn_wmma_f32_16x16x32_bf16(
          false, ap, false, bones, (short)0, osum, false, false);
    }
  }

  // ---- epilogue: O = O / l * outScale; l lives in the nl==0 lane per row.
  float inv[8];
#pragma unroll
  for (int i = 0; i < 8; ++i) {
    float lsum = __shfl(osum[i], lane & 16, 32);   // broadcast column-0 lane
    inv[i] = outScale / lsum;
  }
  const size_t bOff = (size_t)b * Hn * Sq * Dh;
#pragma unroll
  for (int t = 0; t < 4; ++t)
#pragma unroll
    for (int i = 0; i < 8; ++i) {
      int row = qbase + wave * 16 + i + 8 * half;
      int col = t * 16 + nl;
      O[bOff + (size_t)h * oHStr + (size_t)row * oSStr + col] = f2bf(o[t][i] * inv[i]);
    }
}

// ---------------------------------------------------------------------------
extern "C" void kernel_launch(void* const* d_in, const int* in_sizes, int n_in,
                              void* d_out, int out_size, void* d_ws, size_t ws_size,
                              hipStream_t stream) {
  const float* query = (const float*)d_in[0];
  const float* key   = (const float*)d_in[1];
  const float* value = (const float*)d_in[2];
  const float* Wq = (const float*)d_in[3];  const float* bq = (const float*)d_in[4];
  const float* Wk = (const float*)d_in[5];  const float* bk = (const float*)d_in[6];
  const float* Wv = (const float*)d_in[7];  const float* bv = (const float*)d_in[8];
  const float* Wo = (const float*)d_in[9];  const float* bo = (const float*)d_in[10];
  const float* gamma = (const float*)d_in[11];
  const float* beta  = (const float*)d_in[12];
  float* out = (float*)d_out;

  char* ws = (char*)d_ws;
  const size_t fpBytes = (size_t)Mrows * Dm * sizeof(float);          // 16 MB
  const size_t bfBytes = (size_t)Mrows * Dm * sizeof(unsigned short); // 8 MB
  float* Qp = (float*)(ws);
  float* Kp = (float*)(ws + fpBytes);
  float* Vp = (float*)(ws + 2 * fpBytes);
  unsigned short* Qh = (unsigned short*)(ws + 3 * fpBytes);
  unsigned short* Kh = (unsigned short*)(ws + 3 * fpBytes + bfBytes);
  unsigned short* Vh = (unsigned short*)(ws + 3 * fpBytes + 2 * bfBytes);
  // fp32 projection buffers are dead after ln_pack -> alias ping-pong/attn-out
  unsigned short* Q2 = (unsigned short*)Qp;
  unsigned short* Ao = (unsigned short*)Kp;

  // logit scale folded into Q path, exp2-domain softmax
  const float cQ = 0.125f * 1.44269504088896f;   // 1/sqrt(64) * log2(e)

  dim3 gGemm(Dm / 128, Mrows / 128);   // (8, 32)
  // Stage 1: projections (fp32 GEMM + bias via bf16 WMMA)
  gemm_bias_kernel<false><<<gGemm, 256, 0, stream>>>(query, Wq, bq, Qp, Mrows, Dm, Dm);
  gemm_bias_kernel<false><<<gGemm, 256, 0, stream>>>(key,   Wk, bk, Kp, Mrows, Dm, Dm);
  gemm_bias_kernel<false><<<gGemm, 256, 0, stream>>>(value, Wv, bv, Vp, Mrows, Dm, Dm);

  // Stage 2: LayerNorm (q,k) + head-major bf16 repack; v pack only
  int lnBlocks = (Bn * Hn * Sq) / 8;   // 8192
  ln_pack_kernel<<<lnBlocks, 256, 0, stream>>>(Qp, gamma, beta, Qh, 1, cQ);
  ln_pack_kernel<<<lnBlocks, 256, 0, stream>>>(Kp, gamma, beta, Kh, 1, 1.0f);
  ln_pack_kernel<<<lnBlocks, 256, 0, stream>>>(Vp, gamma, beta, Vh, 0, 1.0f);

  // Stage 3: Hopfield iterations (flash attention, V:=K) then readout (V:=V)
  dim3 gAttn(Sq / 128, Hn, Bn);        // (16, 16, 2)
  const long long hsUpd = (long long)Sq * Dh, ssUpd = Dh;          // (b,h,s,d)
  const long long hsFin = Dh, ssFin = (long long)Hn * Dh;          // (b,s,h,d)
  hopfield_attn_kernel<<<gAttn, 256, 0, stream>>>(Qh, Kh, Kh, Q2, hsUpd, ssUpd, cQ);
  hopfield_attn_kernel<<<gAttn, 256, 0, stream>>>(Q2, Kh, Kh, Qh, hsUpd, ssUpd, cQ);
  hopfield_attn_kernel<<<gAttn, 256, 0, stream>>>(Qh, Kh, Kh, Q2, hsUpd, ssUpd, cQ);
  hopfield_attn_kernel<<<gAttn, 256, 0, stream>>>(Q2, Kh, Vh, Ao, hsFin, ssFin, 1.0f);

  // Stage 4: output projection (bf16 A path)
  gemm_bias_kernel<true><<<gGemm, 256, 0, stream>>>(Ao, Wo, bo, out, Mrows, Dm, Dm);
}
